// BinaryClass_5815385719217
// MI455X (gfx1250) — matrix operand order repaired and checked by reading, not compiled, not run on hardware
//
#include <hip/hip_runtime.h>
#include <hip/hip_bf16.h>
#include <math.h>

typedef __attribute__((ext_vector_type(16))) _Float16 v16h;
typedef __attribute__((ext_vector_type(8)))  float    v8f;

#define N_NODES  100000
#define N_EDGES  3200000
#define N_GRAPHS 1000
#define MSG      25                  // T*F = 5*5 messages per edge
#define AVG_LOG  3.4965075614664802f // log(33)

// Accumulator layout: per (node, msg-column) a 16B group {sum, sumsq, mn_enc, mx_enc}
// -> one address per scatter group, immediate offsets for the 4 atomics,
//    and a single b128 load per group on the gather side.

// ---------- helpers ----------
// Order-preserving float<->uint map (branch-free) for exact integer min/max atomics.
__device__ __forceinline__ unsigned f2ord(float f) {
  unsigned u = __float_as_uint(f);
  return u ^ (0x80000000u | (unsigned)((int)u >> 31));
}
__device__ __forceinline__ float ord2f(unsigned e) {
  return (e & 0x80000000u) ? __uint_as_float(e ^ 0x80000000u)
                           : __uint_as_float(~e);
}
__device__ __forceinline__ void atomAddF(float* p, float v) {
  __hip_atomic_fetch_add(p, v, __ATOMIC_RELAXED, __HIP_MEMORY_SCOPE_AGENT);
}
__device__ __forceinline__ void atomMinU(unsigned* p, unsigned v) {
  __hip_atomic_fetch_min(p, v, __ATOMIC_RELAXED, __HIP_MEMORY_SCOPE_AGENT);
}
__device__ __forceinline__ void atomMaxU(unsigned* p, unsigned v) {
  __hip_atomic_fetch_max(p, v, __ATOMIC_RELAXED, __HIP_MEMORY_SCOPE_AGENT);
}

// ---------- init kernels ----------
__global__ void init_accum_kernel(uint4* __restrict__ accum4, float* __restrict__ cnt) {
  int i = blockIdx.x * blockDim.x + threadIdx.x;
  int stride = gridDim.x * blockDim.x;
  const uint4 z = make_uint4(0u, 0u, 0xFFFFFFFFu, 0u); // sum=0, sumsq=0, mn=+inf, mx=-inf
  for (int j = i; j < N_NODES * MSG; j += stride) accum4[j] = z;
  for (int j = i; j < N_NODES; j += stride) cnt[j] = 0.f;
}

__global__ void init_misc_kernel(float* __restrict__ bn_acc, float* __restrict__ gpool) {
  int i = blockIdx.x * blockDim.x + threadIdx.x;
  int stride = gridDim.x * blockDim.x;
  for (int j = i; j < 20; j += stride) bn_acc[j] = 0.f;
  for (int j = i; j < N_GRAPHS * 5; j += stride) gpool[j] = 0.f;
}

// ---------- edge kernel: WMMA message GEMM + atomic segment reduce ----------
// One wave handles 16 edges per tile. A = h[16 x 32] (f16, K padded 15->32),
// B = weights [32 x 16], two N-tiles cover the 25 output columns.
__global__ __launch_bounds__(256)
void pna_edge_kernel(const float* __restrict__ x,
                     const int*   __restrict__ ei,   // [2, E] (row0=src, row1=dst)
                     const float* __restrict__ ea,   // [E]
                     const float* __restrict__ ew,   // [5]  edge encoder weight (1->5)
                     const float* __restrict__ eb,   // [5]
                     const float* __restrict__ pw,   // [T,15,5] pre-MLP weights (this layer)
                     const float* __restrict__ pb,   // [25]
                     float* __restrict__ accum,      // [N, 25, 4] interleaved groups
                     float* __restrict__ cnt) {
  const int tid    = blockIdx.x * blockDim.x + threadIdx.x;
  const int wave   = tid >> 5;
  const int lane   = tid & 31;
  const int nwaves = (gridDim.x * blockDim.x) >> 5;
  const int row    = lane & 15;   // edge row within tile / D column N
  const int hi     = lane >> 4;   // which K-half (A) / which M-half (D)
  const int ntiles = N_EDGES / 16;

  // --- B fragments (uniform given weights). Lanes 0-15 hold K=0..7 (2/VGPR),
  // lanes 16-31 hold K=8..15 (K>=15 zero padding).
  v16h b0 = {}, b1 = {};
  {
#pragma unroll
    for (int j = 0; j < 4; ++j) {
      int k0 = 8 * hi + 2 * j, k1 = 8 * hi + 2 * j + 1;
      int c0 = row, c1 = row + 16;
      float w00 = (k0 < 15) ? pw[((c0 / 5) * 15 + k0) * 5 + (c0 % 5)] : 0.f;
      float w01 = (k1 < 15) ? pw[((c0 / 5) * 15 + k1) * 5 + (c0 % 5)] : 0.f;
      b0[2 * j]     = (_Float16)w00;
      b0[2 * j + 1] = (_Float16)w01;
      if (c1 < MSG) {
        float w10 = (k0 < 15) ? pw[((c1 / 5) * 15 + k0) * 5 + (c1 % 5)] : 0.f;
        float w11 = (k1 < 15) ? pw[((c1 / 5) * 15 + k1) * 5 + (c1 % 5)] : 0.f;
        b1[2 * j]     = (_Float16)w10;
        b1[2 * j + 1] = (_Float16)w11;
      }
    }
  }
  const float bias0 = pb[row];
  const float bias1 = (row < 9) ? pb[row + 16] : 0.f;
  float ewv[5], ebv[5];
#pragma unroll
  for (int j = 0; j < 5; ++j) { ewv[j] = ew[j]; ebv[j] = eb[j]; }

  for (int tile = wave; tile < ntiles; tile += nwaves) {
    const int e = tile * 16 + row;
    const int s = ei[e];
    const int d = ei[N_EDGES + e];

    { // prefetch next tile's edge stream (-> global_prefetch_b8)
      long ne = (long)(tile + nwaves) * 16 + row;
      if (ne < N_EDGES) {
        __builtin_prefetch(ei + ne, 0, 1);
        __builtin_prefetch(ei + N_EDGES + ne, 0, 1);
        __builtin_prefetch(ea + ne, 0, 1);
      }
    }

    // --- A fragment: h = [x_dst(5), x_src(5), edge_enc(5), pad] as f16.
    // Lanes 0-15: K=0..7 ; lanes 16-31: K=8..15 (K>=15 is zero pad).
    v16h a = {};
    if (hi == 0) {
      const float* xd = x + (long)d * 5;
      const float* xs = x + (long)s * 5;
      a[0] = (_Float16)xd[0]; a[1] = (_Float16)xd[1]; a[2] = (_Float16)xd[2];
      a[3] = (_Float16)xd[3]; a[4] = (_Float16)xd[4];
      a[5] = (_Float16)xs[0]; a[6] = (_Float16)xs[1]; a[7] = (_Float16)xs[2];
    } else {
      const float* xs = x + (long)s * 5;
      float eav = ea[e];
      a[0] = (_Float16)xs[3]; a[1] = (_Float16)xs[4];
      a[2] = (_Float16)(eav * ewv[0] + ebv[0]);
      a[3] = (_Float16)(eav * ewv[1] + ebv[1]);
      a[4] = (_Float16)(eav * ewv[2] + ebv[2]);
      a[5] = (_Float16)(eav * ewv[3] + ebv[3]);
      a[6] = (_Float16)(eav * ewv[4] + ebv[4]);
      // a[7] = h[15] = 0 (pad)
    }

    v8f c0 = {}, c1 = {};
    c0 = __builtin_amdgcn_wmma_f32_16x16x32_f16(false, a, false, b0, (short)0, c0, false, false);
    c1 = __builtin_amdgcn_wmma_f32_16x16x32_f16(false, a, false, b1, (short)0, c1, false, false);

    // --- scatter-reduce. D layout: lane holds column N=row, VGPR v -> edge row v+8*hi.
    // One address per group; the 4 atomics use immediate offsets, col+16 is +256B.
#pragma unroll
    for (int v = 0; v < 8; ++v) {
      const int m    = v + 8 * hi;
      const int node = __shfl(d, m, 32);          // dst of edge (tile*16+m)
      float* p = accum + ((long)node * MSG + row) * 4;
      float m0 = c0[v] + bias0;
      unsigned e0 = f2ord(m0);
      atomAddF(p + 0, m0);
      atomAddF(p + 1, m0 * m0);
      atomMinU((unsigned*)p + 2, e0);
      atomMaxU((unsigned*)p + 3, e0);
      if (row < 9) {
        float m1 = c1[v] + bias1;
        unsigned e1 = f2ord(m1);
        atomAddF(p + 64, m1);          // column row+16 -> +16 groups = +64 floats
        atomAddF(p + 65, m1 * m1);
        atomMinU((unsigned*)p + 66, e1);
        atomMaxU((unsigned*)p + 67, e1);
      }
      if (row == 0) atomAddF(&cnt[node], 1.0f);   // exactly once per edge
    }
  }
}

// ---------- node kernel: aggregators + scalers + post-MLP + lin, BN stats ----------
__global__ __launch_bounds__(256)
void pna_node_kernel(const float* __restrict__ x,
                     const uint4* __restrict__ accum4, // [N*25] groups {sum,sumsq,mn,mx}
                     const float* __restrict__ cnt,
                     const float* __restrict__ qw,   // [T,65] post weights (this layer)
                     const float* __restrict__ qb,   // [T]
                     const float* __restrict__ lw,   // [5,5]
                     const float* __restrict__ lb,   // [5]
                     float* __restrict__ xout,       // [N,5] (pre-BN)
                     float* __restrict__ bn_acc) {   // [10]: sum(5), sumsq(5)
  int n = blockIdx.x * blockDim.x + threadIdx.x;
  bool valid = (n < N_NODES);
  float xv[5] = {0.f, 0.f, 0.f, 0.f, 0.f};
  if (valid) {
    float cn = cnt[n];
    float c  = fmaxf(cn, 1.0f);
    float ld = logf(c + 1.0f);
    float amp = ld / AVG_LOG;
    float att = AVG_LOG / ld;
    bool nonempty = (cn > 0.f);
    float xin[5];
#pragma unroll
    for (int i = 0; i < 5; ++i) xin[i] = x[(long)n * 5 + i];
    float y[5];
#pragma unroll
    for (int t = 0; t < 5; ++t) {
      const float* w = qw + t * 65;
      float acc = qb[t];
#pragma unroll
      for (int i = 0; i < 5; ++i) acc += xin[i] * w[i];
#pragma unroll
      for (int f = 0; f < 5; ++f) {
        int j = t * 5 + f;
        uint4 g = accum4[(long)n * MSG + j];     // single b128 load per group
        float s1 = __uint_as_float(g.x);
        float s2 = __uint_as_float(g.y);
        float mean = s1 / c;
        float var  = fmaxf(s2 / c - mean * mean, 0.f);
        float sd   = sqrtf(var + 1e-5f);
        float mnv  = nonempty ? ord2f(g.z) : 0.f;
        float mxv  = nonempty ? ord2f(g.w) : 0.f;
        float ag[4] = {mean, mnv, mxv, sd};
        float sc[3] = {1.f, amp, att};
#pragma unroll
        for (int si = 0; si < 3; ++si)
#pragma unroll
          for (int ai = 0; ai < 4; ++ai)
            acc += ag[ai] * sc[si] * w[5 + si * 20 + ai * 5 + f];
      }
      y[t] = acc;
    }
    // final 5x5 mixing linear
#pragma unroll
    for (int ch = 0; ch < 5; ++ch) {
      float v = lb[ch];
#pragma unroll
      for (int t = 0; t < 5; ++t) v += y[t] * lw[t * 5 + ch];
      xv[ch] = v;
      xout[(long)n * 5 + ch] = v;
    }
  }
  // wave32 reduction of BN statistics, one atomic per wave per channel
#pragma unroll
  for (int ch = 0; ch < 5; ++ch) {
    float s = xv[ch], q = xv[ch] * xv[ch];
#pragma unroll
    for (int off = 16; off; off >>= 1) {
      s += __shfl_xor(s, off, 32);
      q += __shfl_xor(q, off, 32);
    }
    if ((threadIdx.x & 31) == 0) {
      atomAddF(&bn_acc[ch], s);
      atomAddF(&bn_acc[5 + ch], q);
    }
  }
}

// ---------- batchnorm (training-mode biased stats) + ReLU, in place ----------
__global__ void bn_relu_kernel(float* __restrict__ xbuf, const float* __restrict__ bn_acc,
                               const float* __restrict__ g, const float* __restrict__ b) {
  int i = blockIdx.x * blockDim.x + threadIdx.x;
  if (i >= N_NODES * 5) return;
  int ch = i % 5;
  const float inv_n = 1.0f / (float)N_NODES;
  float m = bn_acc[ch] * inv_n;
  float v = bn_acc[5 + ch] * inv_n - m * m;
  float r = rsqrtf(v + 1e-5f);
  float y = g[ch] * (xbuf[i] - m) * r + b[ch];
  xbuf[i] = fmaxf(y, 0.f);
}

// ---------- global add pool ----------
__global__ void pool_kernel(const float* __restrict__ xbuf, const int* __restrict__ batch,
                            float* __restrict__ gpool) {
  int i = blockIdx.x * blockDim.x + threadIdx.x;
  if (i >= N_NODES * 5) return;
  int n = i / 5, ch = i % 5;
  atomAddF(&gpool[(long)batch[n] * 5 + ch], xbuf[i]);
}

// ---------- readout MLP ----------
__global__ void mlp_kernel(const float* __restrict__ gpool,
                           const float* __restrict__ w1, const float* __restrict__ b1,
                           const float* __restrict__ w2, const float* __restrict__ b2,
                           const float* __restrict__ w3, const float* __restrict__ b3,
                           float* __restrict__ out) {
  int gph = blockIdx.x * blockDim.x + threadIdx.x;
  if (gph >= N_GRAPHS) return;
  float gi[5];
#pragma unroll
  for (int i = 0; i < 5; ++i) gi[i] = gpool[(long)gph * 5 + i];
  float h1[5];
#pragma unroll
  for (int j = 0; j < 5; ++j) {
    float v = b1[j];
#pragma unroll
    for (int i = 0; i < 5; ++i) v += gi[i] * w1[i * 5 + j];
    h1[j] = fmaxf(v, 0.f);
  }
  float h2[10];
#pragma unroll
  for (int j = 0; j < 10; ++j) {
    float v = b2[j];
#pragma unroll
    for (int i = 0; i < 5; ++i) v += h1[i] * w2[i * 10 + j];
    h2[j] = fmaxf(v, 0.f);
  }
  float o = b3[0];
#pragma unroll
  for (int i = 0; i < 10; ++i) o += h2[i] * w3[i];
  out[gph] = o;
}

extern "C" void kernel_launch(void* const* d_in, const int* in_sizes, int n_in,
                              void* d_out, int out_size, void* d_ws, size_t ws_size,
                              hipStream_t stream) {
  const float* x   = (const float*)d_in[0];
  const float* ea  = (const float*)d_in[1];
  const float* ew  = (const float*)d_in[2];   // [2,1,5]
  const float* eb  = (const float*)d_in[3];   // [2,5]
  const float* pw  = (const float*)d_in[4];   // [2,5,15,5]
  const float* pb  = (const float*)d_in[5];   // [2,5,5]
  const float* qw  = (const float*)d_in[6];   // [2,5,65,1]
  const float* qb  = (const float*)d_in[7];   // [2,5,1]
  const float* lw  = (const float*)d_in[8];   // [2,5,5]
  const float* lb  = (const float*)d_in[9];   // [2,5]
  const float* bng = (const float*)d_in[10];  // [2,5]
  const float* bnb = (const float*)d_in[11];  // [2,5]
  const float* w1  = (const float*)d_in[12];
  const float* b1  = (const float*)d_in[13];
  const float* w2  = (const float*)d_in[14];
  const float* b2  = (const float*)d_in[15];
  const float* w3  = (const float*)d_in[16];
  const float* b3  = (const float*)d_in[17];
  const int*   ei  = (const int*)d_in[18];    // [2,E]
  const int*   bat = (const int*)d_in[19];    // [N]
  float* out = (float*)d_out;

  // workspace carve-out
  char* ws = (char*)d_ws;
  size_t off = 0;
  auto carve = [&](size_t bytes) -> char* {
    char* p = ws + off;
    off += (bytes + 255) & ~(size_t)255;
    return p;
  };
  float* accum  = (float*)carve((size_t)N_NODES * MSG * 4 * 4); // interleaved groups
  float* cnt    = (float*)carve((size_t)N_NODES * 4);
  float* x1     = (float*)carve((size_t)N_NODES * 5 * 4);
  float* x2     = (float*)carve((size_t)N_NODES * 5 * 4);
  float* bn_acc = (float*)carve(20 * 4);
  float* gpool  = (float*)carve((size_t)N_GRAPHS * 5 * 4);
  (void)ws_size; (void)in_sizes; (void)n_in; (void)out_size;

  const int nodeBlocks = (N_NODES + 255) / 256;
  const int elemBlocks = (N_NODES * 5 + 255) / 256;

  init_accum_kernel<<<1024, 256, 0, stream>>>((uint4*)accum, cnt);
  init_misc_kernel<<<32, 256, 0, stream>>>(bn_acc, gpool);

  // ---- layer 0 ----
  pna_edge_kernel<<<4096, 256, 0, stream>>>(x, ei, ea, ew, eb, pw, pb, accum, cnt);
  pna_node_kernel<<<nodeBlocks, 256, 0, stream>>>(x, (const uint4*)accum, cnt,
                                                  qw, qb, lw, lb, x1, bn_acc);
  bn_relu_kernel<<<elemBlocks, 256, 0, stream>>>(x1, bn_acc, bng, bnb);

  // ---- layer 1 ----
  init_accum_kernel<<<1024, 256, 0, stream>>>((uint4*)accum, cnt);
  pna_edge_kernel<<<4096, 256, 0, stream>>>(x1, ei, ea, ew + 5, eb + 5, pw + 375, pb + 25,
                                            accum, cnt);
  pna_node_kernel<<<nodeBlocks, 256, 0, stream>>>(x1, (const uint4*)accum, cnt,
                                                  qw + 325, qb + 5, lw + 25, lb + 5,
                                                  x2, bn_acc + 10);
  bn_relu_kernel<<<elemBlocks, 256, 0, stream>>>(x2, bn_acc + 10, bng + 5, bnb + 5);

  // ---- pooling + readout ----
  pool_kernel<<<elemBlocks, 256, 0, stream>>>(x2, bat, gpool);
  mlp_kernel<<<(N_GRAPHS + 255) / 256, 256, 0, stream>>>(gpool, w1, b1, w2, b2, w3, b3, out);
}
